// RNNModel_77318001263047
// MI455X (gfx1250) — matrix operand order repaired and checked by reading, not compiled, not run on hardware
//
#include <hip/hip_runtime.h>
#include <hip/hip_bf16.h>

// ---------------------------------------------------------------------------
// 2-layer GRU (TF GRUCell semantics), B=32, T=4096, F=H=128, fp32 in/out.
// Strategy:
//   phase A (parallel): gxcx = x @ [Wg_x | Wc_x] + [bg|bc]  via bf16 WMMA GEMM
//   phase B (sequential): persistent-workgroup scan, recurrent weights in LDS
//                         (bf16, transposed), per-step WMMA for h@Wg_h and
//                         (r*h)@Wc_h, 2 blocks each owning 16 batch rows.
// ---------------------------------------------------------------------------

typedef __bf16 bf16_t;
typedef __attribute__((ext_vector_type(16))) __bf16 v16bf;
typedef __attribute__((ext_vector_type(8)))  __bf16 v8bf;
typedef __attribute__((ext_vector_type(8)))  float  v8f;
typedef __attribute__((ext_vector_type(4)))  float  v4f;

#define WMMA_BF16(a, b, c) \
  __builtin_amdgcn_wmma_f32_16x16x32_bf16(false, (a), false, (b), (short)0, (c), false, false)

#define GRU_B   32
#define GRU_T   4096
#define GRU_H   128
#define GRU_NG  384   // 2H gates + H candidate pre-activations
#define WSTR    136   // padded bf16 LDS row stride (128 + 8) -> bank rotation

// ---- fragment loaders (CDNA5 wave32 WMMA VGPR layouts) ---------------------

// A-matrix 16x32 bf16 frag: lane<16 -> K {0..7,16..23}, lane>=16 -> K {8..15,24..31}
static __device__ inline v16bf fragA_b16(const bf16_t* base, int row0, int rowStride,
                                         int kk, int lane) {
  const int hi = (lane >> 4) & 1;
  const bf16_t* p = base + (row0 + (lane & 15)) * rowStride + kk * 32 + hi * 8;
  v8bf x0 = *(const v8bf*)(p);
  v8bf x1 = *(const v8bf*)(p + 16);
  return __builtin_shufflevector(x0, x1, 0,1,2,3,4,5,6,7,8,9,10,11,12,13,14,15);
}

// A frag sourced from fp32 memory with on-the-fly bf16 conversion
static __device__ inline v16bf fragA_f32(const float* base, int row0, int rowStride,
                                         int kk, int lane) {
  const int hi = (lane >> 4) & 1;
  const float* p = base + (size_t)(row0 + (lane & 15)) * rowStride + kk * 32 + hi * 8;
  v4f a = *(const v4f*)(p);
  v4f b = *(const v4f*)(p + 4);
  v4f c = *(const v4f*)(p + 16);
  v4f d = *(const v4f*)(p + 20);
  v16bf o;
#pragma unroll
  for (int i = 0; i < 4; ++i) {
    o[i]      = (bf16_t)a[i];
    o[4 + i]  = (bf16_t)b[i];
    o[8 + i]  = (bf16_t)c[i];
    o[12 + i] = (bf16_t)d[i];
  }
  return o;
}

// B-matrix 32x16 bf16 frag from W^T stored [N][K] row-major:
// lane<16 -> K {0..7,16..23} of column N=lane, lane>=16 -> K {8..15,24..31} of column N=lane-16
static __device__ inline v16bf fragB_b16(const bf16_t* base, int row0, int rowStride,
                                         int kk, int lane) {
  const int hi = (lane >> 4) & 1;
  const bf16_t* p = base + (row0 + (lane & 15)) * rowStride + kk * 32 + hi * 8;
  v8bf x0 = *(const v8bf*)(p);
  v8bf x1 = *(const v8bf*)(p + 16);
  return __builtin_shufflevector(x0, x1, 0,1,2,3,4,5,6,7,8,9,10,11,12,13,14,15);
}

// ---- kernel 1: pack weights: Wt[n][k] = bf16(W[k][n]), bcat = [bg|bc] ------
__global__ __launch_bounds__(256)
void gru_prep_kernel(const float* __restrict__ Wg, const float* __restrict__ bg,
                     const float* __restrict__ Wc, const float* __restrict__ bc,
                     bf16_t* __restrict__ Wt, float* __restrict__ bcat) {
  const int idx = blockIdx.x * 256 + threadIdx.x;
  if (idx < GRU_NG * GRU_H) {
    const int n = idx >> 7;      // 0..383
    const int k = idx & 127;     // 0..127  (x-part rows of W)
    const float v = (n < 2 * GRU_H) ? Wg[k * (2 * GRU_H) + n]
                                    : Wc[k * GRU_H + (n - 2 * GRU_H)];
    Wt[idx] = (bf16_t)v;         // idx == n*128 + k
  }
  if (idx < GRU_NG) bcat[idx] = (idx < 2 * GRU_H) ? bg[idx] : bc[idx - 2 * GRU_H];
}

// ---- kernel 2: input-projection GEMM: out[(t*32+b)*384+n] = A@Wt^T + bcat --
// A: [M=131072][128] fp32 (row m = b*T + t).  Block tile 128(M) x 64(N), 8 waves.
__global__ __launch_bounds__(256)
void gru_xproj_gemm_kernel(const float* __restrict__ A, const bf16_t* __restrict__ Wt,
                           const float* __restrict__ bcat, float* __restrict__ out) {
  const int lane = threadIdx.x & 31;
  const int wave = threadIdx.x >> 5;
  const int m0 = blockIdx.x * 128 + wave * 16;
  const int n0 = blockIdx.y * 64;

  v8f acc[4];
#pragma unroll
  for (int tn = 0; tn < 4; ++tn) acc[tn] = (v8f){0.f,0.f,0.f,0.f,0.f,0.f,0.f,0.f};

#pragma unroll
  for (int kk = 0; kk < 4; ++kk) {                 // K = 128 = 4 x 32
    v16bf aF = fragA_f32(A, m0, GRU_H, kk, lane);
#pragma unroll
    for (int tn = 0; tn < 4; ++tn) {
      v16bf bF = fragB_b16(Wt, n0 + tn * 16, GRU_H, kk, lane);
      acc[tn] = WMMA_BF16(aF, bF, acc[tn]);
    }
  }

  const int rowAdd = ((lane >> 4) & 1) * 8;
#pragma unroll
  for (int tn = 0; tn < 4; ++tn) {
    const int col = n0 + tn * 16 + (lane & 15);
    const float bias = bcat[col];
#pragma unroll
    for (int r = 0; r < 8; ++r) {
      const int m = m0 + r + rowAdd;
      const int t = m & (GRU_T - 1);
      const int b = m >> 12;                       // T = 4096
      out[(size_t)(t * GRU_B + b) * GRU_NG + col] = acc[tn][r] + bias;
    }
  }
}

// ---- kernel 3: persistent GRU scan (one block per 16 batch rows) -----------
// LDS layout (dynamic):
//   WgT  bf16 [256][WSTR]   gates recurrent weights, transposed
//   WcT  bf16 [128][WSTR]   candidate recurrent weights, transposed
//   hB   bf16 [16][WSTR]    state (bf16 copy for WMMA A)
//   rhB  bf16 [16][WSTR]    r*h (bf16 for WMMA A)
//   hF   f32  [16][128]     state master copy
//   gat  f32  [16][256]     sigmoid gates (r | u)
#define SM_WGT   0
#define SM_WCT   (SM_WGT + 256 * WSTR * 2)        // 69632
#define SM_HB    (SM_WCT + 128 * WSTR * 2)        // 104448
#define SM_RHB   (SM_HB  + 16 * WSTR * 2)         // 108800
#define SM_HF    (SM_RHB + 16 * WSTR * 2)         // 113152
#define SM_GAT   (SM_HF  + 16 * GRU_H * 4)        // 121344
#define SM_TOTAL (SM_GAT + 16 * 256 * 4)          // 137728 bytes

__global__ __launch_bounds__(256)
void gru_scan_kernel(const float* __restrict__ gxcx,   // [(t*32+b)*384]
                     const float* __restrict__ Wgh,    // [128][256] fp32
                     const float* __restrict__ Wch,    // [128][128] fp32
                     float* __restrict__ out) {        // [b][t][128]
  extern __shared__ char smem[];
  bf16_t* WgT = (bf16_t*)(smem + SM_WGT);
  bf16_t* WcT = (bf16_t*)(smem + SM_WCT);
  bf16_t* hB  = (bf16_t*)(smem + SM_HB);
  bf16_t* rhB = (bf16_t*)(smem + SM_RHB);
  float*  hF  = (float*) (smem + SM_HF);
  float*  gat = (float*) (smem + SM_GAT);

  const int tid  = threadIdx.x;
  const int lane = tid & 31;
  const int wave = tid >> 5;
  const int b0   = blockIdx.x * 16;                // batch rows owned by block

  // one-time: transpose + convert recurrent weights into LDS
  for (int i = tid; i < 256 * GRU_H; i += 256) {
    const int n = i >> 7, k = i & 127;
    WgT[n * WSTR + k] = (bf16_t)Wgh[k * 256 + n];
  }
  for (int i = tid; i < GRU_H * GRU_H; i += 256) {
    const int n = i >> 7, k = i & 127;
    WcT[n * WSTR + k] = (bf16_t)Wch[k * GRU_H + n];
  }
  for (int i = tid; i < 16 * GRU_H; i += 256) {
    const int row = i >> 7, n = i & 127;
    hF[i] = 0.f;
    hB[row * WSTR + n] = (bf16_t)0.f;
  }
  __syncthreads();

  const int rowAdd = ((lane >> 4) & 1) * 8;

  for (int t = 0; t < GRU_T; ++t) {
    const float* gx = gxcx + ((size_t)t * GRU_B + b0) * GRU_NG;

    // prefetch next step's pre-activation slab (24 KB) into cache
    if (t + 1 < GRU_T) {
      const char* nxt = (const char*)(gxcx + ((size_t)(t + 1) * GRU_B + b0) * GRU_NG);
      __builtin_prefetch(nxt + tid * 96, 0, 0);
    }

    // --- gates GEMM: G = h @ Wg_h   (16x256, K=128); wave -> n-tiles {2w,2w+1}
    v8f ag0 = (v8f){0.f,0.f,0.f,0.f,0.f,0.f,0.f,0.f};
    v8f ag1 = (v8f){0.f,0.f,0.f,0.f,0.f,0.f,0.f,0.f};
#pragma unroll
    for (int kk = 0; kk < 4; ++kk) {
      v16bf aF  = fragA_b16(hB, 0, WSTR, kk, lane);
      v16bf bF0 = fragB_b16(WgT, wave * 32,      WSTR, kk, lane);
      v16bf bF1 = fragB_b16(WgT, wave * 32 + 16, WSTR, kk, lane);
      ag0 = WMMA_BF16(aF, bF0, ag0);
      ag1 = WMMA_BF16(aF, bF1, ag1);
    }
#pragma unroll
    for (int tn = 0; tn < 2; ++tn) {
      const v8f acc = tn ? ag1 : ag0;
      const int col = wave * 32 + tn * 16 + (lane & 15);
#pragma unroll
      for (int r = 0; r < 8; ++r) {
        const int row = r + rowAdd;
        const float pre = acc[r] + gx[row * GRU_NG + col];
        gat[row * 256 + col] = 1.f / (1.f + __expf(-pre));   // sigmoid
      }
    }
    __syncthreads();

    // --- rh = r * h  (bf16, WMMA-A layout)
#pragma unroll
    for (int i = 0; i < 8; ++i) {
      const int idx = tid + i * 256;                 // 0..2047
      const int row = idx >> 7, n = idx & 127;
      rhB[row * WSTR + n] = (bf16_t)(gat[row * 256 + n] * hF[row * GRU_H + n]);
    }
    __syncthreads();

    // --- candidate GEMM: C = (r*h) @ Wc_h  (16x128); wave -> n-tile w
    v8f ac = (v8f){0.f,0.f,0.f,0.f,0.f,0.f,0.f,0.f};
#pragma unroll
    for (int kk = 0; kk < 4; ++kk) {
      v16bf aF = fragA_b16(rhB, 0, WSTR, kk, lane);
      v16bf bF = fragB_b16(WcT, wave * 16, WSTR, kk, lane);
      ac = WMMA_BF16(aF, bF, ac);
    }
    {
      const int col = wave * 16 + (lane & 15);
#pragma unroll
      for (int r = 0; r < 8; ++r) {
        const int row = r + rowAdd;
        const float pre = ac[r] + gx[row * GRU_NG + 2 * GRU_H + col];
        const float e = __expf(2.f * pre);
        const float c = 1.f - 2.f / (e + 1.f);       // tanh
        const float u = gat[row * 256 + GRU_H + col];
        const float ho = hF[row * GRU_H + col];
        const float hn = u * ho + (1.f - u) * c;
        hF[row * GRU_H + col] = hn;
        hB[row * WSTR + col]  = (bf16_t)hn;
        out[((size_t)(b0 + row) * GRU_T + t) * GRU_H + col] = hn;
      }
    }
    __syncthreads();
  }
}

// ---------------------------------------------------------------------------
extern "C" void kernel_launch(void* const* d_in, const int* in_sizes, int n_in,
                              void* d_out, int out_size, void* d_ws, size_t ws_size,
                              hipStream_t stream) {
  (void)in_sizes; (void)n_in; (void)out_size; (void)ws_size;

  const float* x   = (const float*)d_in[0];
  const float* Wg0 = (const float*)d_in[1];
  const float* bg0 = (const float*)d_in[2];
  const float* Wc0 = (const float*)d_in[3];
  const float* bc0 = (const float*)d_in[4];
  const float* Wg1 = (const float*)d_in[5];
  const float* bg1 = (const float*)d_in[6];
  const float* Wc1 = (const float*)d_in[7];
  const float* bc1 = (const float*)d_in[8];

  // workspace carve-up
  char* ws = (char*)d_ws;
  bf16_t* Wt   = (bf16_t*)(ws);                       //  98,304 B
  float*  bcat = (float*) (ws + 98304);               //   1,536 B
  float*  gxcx = (float*) (ws + 102400);              // 201,326,592 B (reused per layer)
  float*  h1   = (float*) (ws + 102400 + 201326592ull); // 67,108,864 B

  static int smem_attr_set_done = 0; // harmless: setting the same value is idempotent & deterministic
  (void)smem_attr_set_done;
  hipFuncSetAttribute((const void*)gru_scan_kernel,
                      hipFuncAttributeMaxDynamicSharedMemorySize, SM_TOTAL);

  const dim3 gemmGrid((GRU_B * GRU_T) / 128, GRU_NG / 64);

  // ---- layer 0 ----
  gru_prep_kernel<<<192, 256, 0, stream>>>(Wg0, bg0, Wc0, bc0, Wt, bcat);
  gru_xproj_gemm_kernel<<<gemmGrid, 256, 0, stream>>>(x, Wt, bcat, gxcx);
  gru_scan_kernel<<<2, 256, SM_TOTAL, stream>>>(gxcx,
                                                Wg0 + 128 * 256,   // Wg_h
                                                Wc0 + 128 * 128,   // Wc_h
                                                h1);

  // ---- layer 1 ----
  gru_prep_kernel<<<192, 256, 0, stream>>>(Wg1, bg1, Wc1, bc1, Wt, bcat);
  gru_xproj_gemm_kernel<<<gemmGrid, 256, 0, stream>>>(h1, Wt, bcat, gxcx);
  gru_scan_kernel<<<2, 256, SM_TOTAL, stream>>>(gxcx,
                                                Wg1 + 128 * 256,
                                                Wc1 + 128 * 128,
                                                (float*)d_out);
}
